// LSTM_custom_85959475462558
// MI455X (gfx1250) — matrix operand order repaired and checked by reading, not compiled, not run on hardware
//
#include <hip/hip_runtime.h>
#include <math.h>

// ---------------------------------------------------------------------------
// 2-layer LSTM for MI455X (gfx1250), wave32 + WMMA bf16 + async LDS staging.
//
// Persistent per-layer kernel: 16 workgroups (= one max-size workgroup
// cluster, one WGP each); WG w owns hidden slice j in [64w, 64w+64).
// Per timestep, the 64x256 gate slab  g = h @ W^T  is computed with
// v_wmma_f32_16x16x32_bf16. Weights (bf16, 24 MB total -> L2-resident) and
// the h slab are staged per 256-wide K-chunk into LDS with
// global_load_async_to_lds_b128 (ASYNCcnt-tracked, VGPR-bypassing; plain
// load+ds_store fallback if the builtin is absent), so each unique byte
// crosses the WGP boundary once instead of 4x. Fragment reads then come from
// ds_load_b128 with an 8-element row pad (528 B pitch -> 4-bank row stride)
// keeping the 16-lane column reads conflict-free. All 8 B fragments of a
// k-step are loaded into distinct registers before the WMMA chain so DScnt
// partial waits can overlap LDS loads with the matrix pipe.
// Cell state c lives in VGPRs for all 512 steps; h is double-buffered in
// global memory and published per step via the CDNA5 cluster split-barrier
// (s_barrier_signal/-wait -3) backed by an agent-scope atomic barrier so the
// kernel is also correct under a plain (non-cluster) dispatch.
// ---------------------------------------------------------------------------

#define HIDDEN 1024
#define GATES  (4 * HIDDEN)
#define BATCH  64
#define SEQ    512
#define NBLK   16            // workgroups per layer kernel == cluster size
#define KC     256           // K-chunk staged in LDS per iteration
#define KCP    (KC + 8)      // padded row pitch (bank-conflict avoidance)
#define SMEM_BYTES ((64 * KCP + 256 * KCP) * 2)   // 168,960 B < 320 KB WGP LDS

typedef __attribute__((ext_vector_type(16))) __bf16 v16bf;
typedef __attribute__((ext_vector_type(8)))  __bf16 v8bf;
typedef __attribute__((ext_vector_type(8)))  float  v8f;
typedef int v4i_b128 __attribute__((vector_size(16)));  // builtin payload type

#if __has_builtin(__builtin_amdgcn_global_load_async_to_lds_b128)
#define USE_ASYNC_LDS 1
#else
#define USE_ASYNC_LDS 0
#endif

__device__ __forceinline__ v16bf cat16(v8bf lo, v8bf hi) {
  return __builtin_shufflevector(lo, hi, 0, 1, 2, 3, 4, 5, 6, 7,
                                 8, 9, 10, 11, 12, 13, 14, 15);
}

__device__ __forceinline__ float sigf(float x) {
  return 1.0f / (1.0f + __expf(-x));
}

// 16-byte global -> LDS copy. Async path: ASYNCcnt-tracked DMA that bypasses
// VGPRs entirely (CDNA5 ISA). Fallback: load + ds_store.
// Builtin signature (from toolchain probe): (v4i AS1*, v4i AS3*, Ii, Ii).
__device__ __forceinline__ void cp16(const __bf16* __restrict__ g,
                                     __bf16* __restrict__ l) {
#if USE_ASYNC_LDS
  __builtin_amdgcn_global_load_async_to_lds_b128(
      (__attribute__((address_space(1))) v4i_b128*)(void*)g,
      (__attribute__((address_space(3))) v4i_b128*)(void*)l, 0, 0);
#else
  *(v8bf*)l = *(const v8bf*)g;
#endif
}

__device__ __forceinline__ void wait_stage() {
#if USE_ASYNC_LDS
#if __has_builtin(__builtin_amdgcn_s_wait_asynccnt)
  __builtin_amdgcn_s_wait_asynccnt(0);
#else
  asm volatile("s_wait_asynccnt 0x0" ::: "memory");
#endif
#endif
  __syncthreads();   // publish staged LDS to all waves in the WG
}

// Grid-wide barrier across the 16 persistent workgroups.
// Real guarantee: agent-scope atomic sense barrier (16 blocks co-resident).
// CDNA5 path: cluster split-barrier, a NOP when ClusterID==0.
__device__ __forceinline__ void grid_barrier(unsigned* cnt, unsigned* gen) {
  __syncthreads();
  if (threadIdx.x == 0) {
    __threadfence();
    unsigned g = __hip_atomic_load(gen, __ATOMIC_ACQUIRE, __HIP_MEMORY_SCOPE_AGENT);
    unsigned old = __hip_atomic_fetch_add(cnt, 1u, __ATOMIC_ACQ_REL, __HIP_MEMORY_SCOPE_AGENT);
    if (old == NBLK - 1) {
      __hip_atomic_store(cnt, 0u, __ATOMIC_RELAXED, __HIP_MEMORY_SCOPE_AGENT);
      __hip_atomic_fetch_add(gen, 1u, __ATOMIC_ACQ_REL, __HIP_MEMORY_SCOPE_AGENT);
    } else {
      while (__hip_atomic_load(gen, __ATOMIC_ACQUIRE, __HIP_MEMORY_SCOPE_AGENT) == g)
        __builtin_amdgcn_s_sleep(1);
    }
    __threadfence();
  }
  __builtin_amdgcn_s_cluster_barrier();   // s_barrier_signal -3 ; s_barrier_wait -3
  __syncthreads();
}

// Stage one K-chunk into LDS:
//   sA[64][KCP]  <- 64 rows of the A matrix  (abase + m*astride + kc ...)
//   sB[256][KCP] <- 256 weight rows: local col c -> global row
//                   (c>>6)*HIDDEN + jBase + (c&63)  (4 gates x 64 units)
__device__ __forceinline__ void stage_chunk(const __bf16* __restrict__ abase,
                                            size_t astride,
                                            const __bf16* __restrict__ W,
                                            int jBase, int kc,
                                            __bf16* __restrict__ sA,
                                            __bf16* __restrict__ sB, int tid) {
#pragma unroll
  for (int it = 0; it < (64 * KC / 8) / 256; ++it) {     // 8 iters
    const int idx = tid + it * 256;
    const int m = idx >> 5, vv = (idx & 31) * 8;
    cp16(abase + (size_t)m * astride + kc + vv, sA + m * KCP + vv);
  }
#pragma unroll
  for (int it = 0; it < (256 * KC / 8) / 256; ++it) {    // 32 iters
    const int idx = tid + it * 256;
    const int c = idx >> 5, vv = (idx & 31) * 8;
    const __bf16* src =
        W + (size_t)((c >> 6) * HIDDEN + jBase + (c & 63)) * HIDDEN + kc + vv;
    __builtin_prefetch(src + KC, 0, 3);                  // global_prefetch_b8
    cp16(src, sB + c * KCP + vv);
  }
}

// acc[jt][gate] += A(16x32) x B(32x16) over the staged KC chunk. All operands
// come from LDS; all 8 B fragments are loaded into independent registers
// before the WMMA chain so the compiler can issue partial s_wait_dscnt and
// overlap ds_load_b128 with the matrix pipe.
__device__ __forceinline__ void gemm_chunk(const __bf16* __restrict__ sA,
                                           const __bf16* __restrict__ sB,
                                           int mBase, int jt0, int laneN,
                                           int aK0, int bK0, v8f acc[2][4]) {
  const __bf16* arow = sA + (mBase + laneN) * KCP;
#pragma unroll 2
  for (int kk = 0; kk < KC; kk += 32) {
    v16bf a = cat16(*(const v8bf*)(arow + kk + aK0),
                    *(const v8bf*)(arow + kk + aK0 + 16));
    v16bf b[8];
#pragma unroll
    for (int g = 0; g < 4; ++g)
#pragma unroll
      for (int jt = 0; jt < 2; ++jt) {
        const __bf16* brow =
            sB + (g * 64 + (jt0 + jt) * 16 + laneN) * KCP + kk + bK0;
        b[g * 2 + jt] = cat16(*(const v8bf*)brow, *(const v8bf*)(brow + 16));
      }
#pragma unroll
    for (int g = 0; g < 4; ++g)
#pragma unroll
      for (int jt = 0; jt < 2; ++jt)
        acc[jt][g] = __builtin_amdgcn_wmma_f32_16x16x32_bf16(
            false, a, false, b[g * 2 + jt], (short)0, acc[jt][g], false,
            false);
  }
}

// Full K=HIDDEN pass: 4 staged chunks.
__device__ __forceinline__ void gemm_pass(const __bf16* __restrict__ abase,
                                          size_t astride,
                                          const __bf16* __restrict__ W,
                                          int jBase, __bf16* sA, __bf16* sB,
                                          int tid, int mBase, int jt0,
                                          int laneN, int aK0, int bK0,
                                          v8f acc[2][4]) {
  for (int kc = 0; kc < HIDDEN; kc += KC) {
    __syncthreads();                       // previous chunk fully consumed
    stage_chunk(abase, astride, W, jBase, kc, sA, sB, tid);
    wait_stage();                          // s_wait_asynccnt 0 + barrier
    gemm_chunk(sA, sB, mBase, jt0, laneN, aK0, bK0, acc);
  }
}

// ---------------------------------------------------------------------------
// Persistent LSTM layer kernel. 16 blocks x 256 threads (8 wave32 per block).
// Wave w: mTile = w>>1 (batch rows 16*mTile..+16), jTiles {2*(w&1), 2*(w&1)+1}
// within the WG's 64-wide hidden slice. Each wave keeps 4 gate accumulators
// per jTile -> i/f/g/o for (m, j) land in identical lane/VGPR slots, so the
// LSTM pointwise update is entirely lane-local.
// ---------------------------------------------------------------------------
template <int LAYER>
__global__ __launch_bounds__(256, 1) void lstm_layer_kernel(
    const __bf16* __restrict__ Whh,   // [4H][H] bf16
    const __bf16* __restrict__ Wih,   // layer1: [4H][H] bf16 (layer0: unused)
    const float* __restrict__ x0,     // layer0: data [B][T] (IN==1)
    const float* __restrict__ wih0,   // layer0: W_ih0 [4H]
    const float* __restrict__ b_ih,   // [4H]
    const float* __restrict__ b_hh,   // [4H]
    const __bf16* __restrict__ Xb,    // layer1: h1 bf16 [B][T][H]
    __bf16* __restrict__ Hst,         // [2][B][H] bf16 ping-pong h state
    __bf16* __restrict__ H1out,       // layer0: [B][T][H] bf16
    float* __restrict__ Out,          // layer1: [B][T][H] f32 (d_out)
    unsigned* __restrict__ bar_cnt, unsigned* __restrict__ bar_gen) {
  extern __shared__ __bf16 smem[];
  __bf16* sA = smem;                 // [64][KCP]
  __bf16* sB = smem + 64 * KCP;      // [256][KCP]

  const int tid = threadIdx.x;
  const int wave = tid >> 5;
  const int lane = tid & 31;
  const int laneN = lane & 15;
  const int laneHalf = lane >> 4;
  const int jBase = blockIdx.x * 64;   // this WG's hidden slice
  const int mBase = (wave >> 1) * 16;  // batch-row tile
  const int jt0 = (wave & 1) * 2;

  int jj[2];
  jj[0] = jBase + (jt0 + 0) * 16 + laneN;
  jj[1] = jBase + (jt0 + 1) * 16 + laneN;

  // Per-lane constants: combined bias (+ scalar input weight for layer 0).
  float bias[2][4], wi0[2][4];
#pragma unroll
  for (int jt = 0; jt < 2; ++jt)
#pragma unroll
    for (int g = 0; g < 4; ++g) {
      const int gi = g * HIDDEN + jj[jt];
      bias[jt][g] = b_ih[gi] + b_hh[gi];
      wi0[jt][g] = (LAYER == 0) ? wih0[gi] : 0.0f;
    }

  float cst[2][8] = {};  // cell state, resident in VGPRs for all T steps

  // Zero this WG's slice of the t=0 h buffer.
  for (int idx = tid; idx < 64 * 64; idx += 256) {
    const int m = idx >> 6, j = idx & 63;
    Hst[(size_t)m * HIDDEN + jBase + j] = (__bf16)0.0f;
  }
  grid_barrier(bar_cnt, bar_gen);

  const int aK0 = laneHalf * 8;     // ISA 16-bit A layout: K chunks {base, base+16}
  const int bK0 = laneHalf * 8;     // B layout matches A: K chunks {base, base+16}

  for (int t = 0; t < SEQ; ++t) {
    const __bf16* hbuf = Hst + (size_t)(t & 1) * (BATCH * HIDDEN);
    __bf16* hnext = Hst + (size_t)((t + 1) & 1) * (BATCH * HIDDEN);

    v8f acc[2][4];
#pragma unroll
    for (int jt = 0; jt < 2; ++jt)
#pragma unroll
      for (int g = 0; g < 4; ++g)
        acc[jt][g] = (v8f){0.f, 0.f, 0.f, 0.f, 0.f, 0.f, 0.f, 0.f};

    // Recurrent term: h_{t-1} @ W_hh^T (A rows have stride HIDDEN).
    gemm_pass(hbuf, (size_t)HIDDEN, Whh, jBase, sA, sB, tid, mBase, jt0,
              laneN, aK0, bK0, acc);
    // Layer-1 input term as a second K pass: x_t @ W_ih^T
    // (row m of x_t sits at Xb + (m*SEQ + t)*HIDDEN -> stride SEQ*HIDDEN).
    if (LAYER == 1)
      gemm_pass(Xb + (size_t)t * HIDDEN, (size_t)SEQ * HIDDEN, Wih, jBase, sA,
                sB, tid, mBase, jt0, laneN, aK0, bK0, acc);

    // Gate nonlinearities + state update (all lane-local by construction).
#pragma unroll
    for (int jt = 0; jt < 2; ++jt) {
#pragma unroll
      for (int r = 0; r < 8; ++r) {
        const int m = mBase + laneHalf * 8 + r;  // C/D layout: M = half*8 + vgpr
        float gi = acc[jt][0][r] + bias[jt][0];
        float gf = acc[jt][1][r] + bias[jt][1];
        float gg = acc[jt][2][r] + bias[jt][2];
        float go = acc[jt][3][r] + bias[jt][3];
        if (LAYER == 0) {  // IN==1: input GEMM degenerates to scalar * vector
          const float xv = x0[m * SEQ + t];
          gi = fmaf(xv, wi0[jt][0], gi);
          gf = fmaf(xv, wi0[jt][1], gf);
          gg = fmaf(xv, wi0[jt][2], gg);
          go = fmaf(xv, wi0[jt][3], go);
        }
        gi = sigf(gi);
        gf = sigf(gf);
        gg = tanhf(gg);
        go = sigf(go);
        const float cn = fmaf(gf, cst[jt][r], gi * gg);
        cst[jt][r] = cn;
        const float hn = go * tanhf(cn);
        hnext[(size_t)m * HIDDEN + jj[jt]] = (__bf16)hn;
        if (LAYER == 0)
          H1out[((size_t)m * SEQ + t) * HIDDEN + jj[jt]] = (__bf16)hn;
        else
          Out[((size_t)m * SEQ + t) * HIDDEN + jj[jt]] = hn;
      }
    }

    grid_barrier(bar_cnt, bar_gen);  // publish h_t before anyone reads it
  }
}

// fp32 -> bf16 weight repack (one-time per launch, deterministic).
__global__ void pack_bf16_kernel(const float* __restrict__ src,
                                 __bf16* __restrict__ dst, int n) {
  for (int i = blockIdx.x * blockDim.x + threadIdx.x; i < n;
       i += gridDim.x * blockDim.x)
    dst[i] = (__bf16)src[i];
}

// readout = output[:, -1, :] @ fc_w^T + fc_b   ([64,1024] x [1024,2])
__global__ void readout_kernel(const float* __restrict__ Out,
                               const float* __restrict__ fcw,
                               const float* __restrict__ fcb,
                               float* __restrict__ ro) {
  const int t = threadIdx.x;  // 128 threads: (b, cls)
  if (t >= BATCH * 2) return;
  const int b = t >> 1, cls = t & 1;
  const float* row = Out + ((size_t)b * SEQ + (SEQ - 1)) * HIDDEN;
  const float* w = fcw + cls * HIDDEN;
  float s = fcb[cls];
  for (int k = 0; k < HIDDEN; ++k) s = fmaf(row[k], w[k], s);
  ro[b * 2 + cls] = s;
}

extern "C" void kernel_launch(void* const* d_in, const int* in_sizes, int n_in,
                              void* d_out, int out_size, void* d_ws,
                              size_t ws_size, hipStream_t stream) {
  const float* data = (const float*)d_in[0];   // [64,512,1]
  const float* W_ih0 = (const float*)d_in[1];  // [4096,1]
  const float* W_hh0 = (const float*)d_in[2];  // [4096,1024]
  const float* b_ih0 = (const float*)d_in[3];
  const float* b_hh0 = (const float*)d_in[4];
  const float* W_ih1 = (const float*)d_in[5];  // [4096,1024]
  const float* W_hh1 = (const float*)d_in[6];  // [4096,1024]
  const float* b_ih1 = (const float*)d_in[7];
  const float* b_hh1 = (const float*)d_in[8];
  const float* fc_w = (const float*)d_in[9];   // [2,1024]
  const float* fc_b = (const float*)d_in[10];  // [2]

  float* Out = (float*)d_out;                      // [64,512,1024]
  float* ro = Out + (size_t)BATCH * SEQ * HIDDEN;  // [64,2]

  // Workspace carve-up (~88.5 MB): bf16 weights resident in the 192 MB L2.
  const size_t WN = (size_t)GATES * HIDDEN;  // 4M elements per weight matrix
  __bf16* Whh0b = (__bf16*)d_ws;
  __bf16* Wih1b = Whh0b + WN;
  __bf16* Whh1b = Wih1b + WN;
  __bf16* H1 = Whh1b + WN;                           // [64,512,1024] bf16
  __bf16* Hst0 = H1 + (size_t)BATCH * SEQ * HIDDEN;  // [2,64,1024] bf16
  __bf16* Hst1 = Hst0 + 2 * BATCH * HIDDEN;          // [2,64,1024] bf16
  unsigned* bars = (unsigned*)(Hst1 + 2 * BATCH * HIDDEN);  // 4 uints

  // 165 KB dynamic LDS per block (WGP budget is 320 KB) — raise the cap.
  (void)hipFuncSetAttribute((const void*)&lstm_layer_kernel<0>,
                            hipFuncAttributeMaxDynamicSharedMemorySize,
                            SMEM_BYTES);
  (void)hipFuncSetAttribute((const void*)&lstm_layer_kernel<1>,
                            hipFuncAttributeMaxDynamicSharedMemorySize,
                            SMEM_BYTES);

  (void)hipMemsetAsync(bars, 0, 4 * sizeof(unsigned), stream);

  pack_bf16_kernel<<<1024, 256, 0, stream>>>(W_hh0, Whh0b, (int)WN);
  pack_bf16_kernel<<<1024, 256, 0, stream>>>(W_ih1, Wih1b, (int)WN);
  pack_bf16_kernel<<<1024, 256, 0, stream>>>(W_hh1, Whh1b, (int)WN);

  lstm_layer_kernel<0><<<NBLK, 256, SMEM_BYTES, stream>>>(
      Whh0b, nullptr, data, W_ih0, b_ih0, b_hh0, nullptr, Hst0, H1, nullptr,
      bars + 0, bars + 1);
  lstm_layer_kernel<1><<<NBLK, 256, SMEM_BYTES, stream>>>(
      Whh1b, Wih1b, nullptr, nullptr, b_ih1, b_hh1, H1, Hst1, nullptr, Out,
      bars + 2, bars + 3);

  readout_kernel<<<1, 128, 0, stream>>>(Out, fc_w, fc_b, ro);
}
